// LocalitySelfAttention_10075993277124
// MI455X (gfx1250) — matrix operand order repaired and checked by reading, not compiled, not run on hardware
//
#include <hip/hip_runtime.h>
#include <hip/hip_bf16.h>

// ---------------------------------------------------------------------------
// LocalitySelfAttention for MI455X (gfx1250), wave32 + BF16 WMMA.
// D = A x B + C via v_wmma_f32_16x16x32_bf16 everywhere; flash-attention
// online softmax in fp32; diagonal self-mask; raw-reshape head layout means
// Q/K/O buffers are layout-identical to the row-major activations.
// GEMM wave tile = 32x64 (2 A-frags share 4 B-frags -> 1.5 loads/WMMA).
// ---------------------------------------------------------------------------

typedef __attribute__((ext_vector_type(16))) __bf16 bf16x16;
typedef __attribute__((ext_vector_type(8)))  __bf16 bf16x8;
typedef __attribute__((ext_vector_type(8)))  float  f32x8;

#define B_    2
#define N_    4096
#define DIM_  512
#define H_    8
#define HD_   64
#define QKVC_ 1536   // 3 * INNER

__device__ __forceinline__ f32x8 wmma_bf16(bf16x16 a, bf16x16 b, f32x8 c) {
  // (neg_a, A, neg_b, B, c_mod, C, reuse_a, reuse_b)
  return __builtin_amdgcn_wmma_f32_16x16x32_bf16(false, a, false, b, (short)0, c,
                                                 false, false);
}

union AFrag { bf16x16 v; bf16x8 h[2]; };

// ---------------------------------------------------------------------------
// Kernel 0: fp32 -> bf16 conversion of x; transposed bf16 copies of weights
// so GEMM B-fragments are 32B-contiguous per lane (K along fastest dim).
// ---------------------------------------------------------------------------
__global__ __launch_bounds__(256) void prep_kernel(
    const float* __restrict__ x, const float* __restrict__ wqkv,
    const float* __restrict__ wout,
    __bf16* __restrict__ xbf, __bf16* __restrict__ wqkvT,
    __bf16* __restrict__ woutT)
{
  const int tid    = blockIdx.x * blockDim.x + threadIdx.x;
  const int stride = gridDim.x * blockDim.x;

  for (int j = tid; j < B_ * N_ * DIM_; j += stride)
    xbf[j] = (__bf16)x[j];

  for (int j = tid; j < QKVC_ * DIM_; j += stride) {
    const int n = j >> 9, k = j & 511;           // wqkvT[n][k] = wqkv[k][n]
    wqkvT[j] = (__bf16)wqkv[(size_t)k * QKVC_ + n];
  }
  for (int j = tid; j < DIM_ * DIM_; j += stride) {
    const int n = j >> 9, k = j & 511;           // woutT[n][k] = wout[k][n]
    woutT[j] = (__bf16)wout[(size_t)k * DIM_ + n];
  }
}

// ---------------------------------------------------------------------------
// Kernel 1: QKV = Xbf (4096x512) @ Wqkv (512x1536), per batch.
// Block = 256 thr = 8 waves; block tile 128(M) x 128(N); wave tile 32x64.
// Epilogue scatters Q/K in flat layout (== head layout under raw reshape)
// and V transposed as Vt[b][h][d][m] for contiguous PV B-fragments.
// ---------------------------------------------------------------------------
__global__ __launch_bounds__(256) void qkv_gemm(
    const __bf16* __restrict__ X, const __bf16* __restrict__ Wt,
    __bf16* __restrict__ Qb, __bf16* __restrict__ Kb, __bf16* __restrict__ Vt)
{
  const int lane = threadIdx.x & 31;
  const int wave = threadIdx.x >> 5;
  const int lo = lane & 15, hi = lane >> 4;
  const int b = blockIdx.z;
  const int row0    = blockIdx.x * 128 + (wave & 3) * 32;
  const int colbase = blockIdx.y * 128 + (wave >> 2) * 64;
  const __bf16* xr0 = X + ((size_t)b * N_ + row0 + lo) * DIM_;
  const __bf16* xr1 = X + ((size_t)b * N_ + row0 + 16 + lo) * DIM_;

  f32x8 acc[2][4];
#pragma unroll
  for (int i = 0; i < 2; ++i)
#pragma unroll
    for (int nt = 0; nt < 4; ++nt)
#pragma unroll
      for (int r = 0; r < 8; ++r) acc[i][nt][r] = 0.0f;

#pragma unroll 2
  for (int kc = 0; kc < DIM_; kc += 32) {
    AFrag a0, a1;                                   // A: 2 x (16x32) bf16
    a0.h[0] = *(const bf16x8*)(xr0 + kc + hi * 16);
    a0.h[1] = *(const bf16x8*)(xr0 + kc + 8 + hi * 16);
    a1.h[0] = *(const bf16x8*)(xr1 + kc + hi * 16);
    a1.h[1] = *(const bf16x8*)(xr1 + kc + 8 + hi * 16);
#pragma unroll
    for (int nt = 0; nt < 4; ++nt) {                // B: 32x16 bf16, contiguous
      bf16x16 bf = *(const bf16x16*)(Wt + (size_t)(colbase + nt * 16 + lo) * DIM_
                                        + kc + hi * 16);
      acc[0][nt] = wmma_bf16(a0.v, bf, acc[0][nt]);
      acc[1][nt] = wmma_bf16(a1.v, bf, acc[1][nt]);
    }
  }

#pragma unroll
  for (int i = 0; i < 2; ++i)
#pragma unroll
  for (int nt = 0; nt < 4; ++nt) {
    const int c = colbase + nt * 16 + lo;
#pragma unroll
    for (int r = 0; r < 8; ++r) {
      const int n = row0 + i * 16 + r + hi * 8;
      const __bf16 v = (__bf16)acc[i][nt][r];
      if (c < 512) {
        Qb[((size_t)b * N_ + n) * 512 + c] = v;
      } else if (c < 1024) {
        Kb[((size_t)b * N_ + n) * 512 + (c - 512)] = v;
      } else {
        const int cc = c - 1024;
        const int h = n >> 9, d = cc & 63, m = ((n & 511) << 3) | (cc >> 6);
        Vt[(((size_t)b * H_ + h) * HD_ + d) * (size_t)N_ + m] = v;
      }
    }
  }
}

// ---------------------------------------------------------------------------
// Kernel 2: flash attention per (b,h). 8 waves/block, 16 query rows/wave,
// key sweep in blocks of 32 keys. Online softmax in fp32; P transposed
// C-layout -> A-layout through per-wave LDS tiles.
// ---------------------------------------------------------------------------
__global__ __launch_bounds__(256) void attn_kernel(
    const __bf16* __restrict__ Qb, const __bf16* __restrict__ Kb,
    const __bf16* __restrict__ Vt, const float* __restrict__ temperature,
    __bf16* __restrict__ Ob)
{
  __shared__ __align__(16) __bf16 P[8][16][40];   // per-wave 16x32 (+pad)

  const int lane = threadIdx.x & 31;
  const int wave = threadIdx.x >> 5;
  const int lo = lane & 15, hi = lane >> 4;
  const int b = blockIdx.z, h = blockIdx.y;
  const int q0 = blockIdx.x * 128 + wave * 16;

  const size_t headOff = ((size_t)b * H_ + h) * (size_t)N_ * HD_;
  const __bf16* Qh  = Qb + headOff;
  const __bf16* Kh  = Kb + headOff;
  const __bf16* Vth = Vt + ((size_t)b * H_ + h) * (size_t)HD_ * N_;

  const float L2E   = 1.44269504f;
  const float scale = exp2f(temperature[0] * L2E);  // exp(temperature)

  // Q A-fragments for K-chunks d[0..31] and d[32..63], loaded once.
  AFrag qa0, qa1;
  {
    const __bf16* qp = Qh + (size_t)(q0 + lo) * HD_;
    qa0.h[0] = *(const bf16x8*)(qp + hi * 16);
    qa0.h[1] = *(const bf16x8*)(qp + 8 + hi * 16);
    qa1.h[0] = *(const bf16x8*)(qp + 32 + hi * 16);
    qa1.h[1] = *(const bf16x8*)(qp + 40 + hi * 16);
  }

  f32x8 O[4];
  float mi[8], li[8];
#pragma unroll
  for (int nt = 0; nt < 4; ++nt)
#pragma unroll
    for (int r = 0; r < 8; ++r) O[nt][r] = 0.0f;
#pragma unroll
  for (int r = 0; r < 8; ++r) { mi[r] = -3.0e38f; li[r] = 0.0f; }

  for (int kb = 0; kb < N_; kb += 32) {
    if (kb + 32 < N_) {                               // global_prefetch_b8
      __builtin_prefetch(Kh + (size_t)(kb + 32 + lo) * HD_, 0, 3);
      __builtin_prefetch(Vth + (size_t)lo * N_ + kb + 32, 0, 3);
    }

    f32x8 s0, s1;
#pragma unroll
    for (int r = 0; r < 8; ++r) { s0[r] = 0.0f; s1[r] = 0.0f; }

    // S = Q Kt : two 16-key tiles, K(dim d)=64 split into 2 chunks of 32.
    s0 = wmma_bf16(qa0.v, *(const bf16x16*)(Kh + (size_t)(kb + lo) * HD_ + hi * 16), s0);
    s0 = wmma_bf16(qa1.v, *(const bf16x16*)(Kh + (size_t)(kb + lo) * HD_ + 32 + hi * 16), s0);
    s1 = wmma_bf16(qa0.v, *(const bf16x16*)(Kh + (size_t)(kb + 16 + lo) * HD_ + hi * 16), s1);
    s1 = wmma_bf16(qa1.v, *(const bf16x16*)(Kh + (size_t)(kb + 16 + lo) * HD_ + 32 + hi * 16), s1);

#pragma unroll
    for (int r = 0; r < 8; ++r) { s0[r] *= scale; s1[r] *= scale; }

    // Diagonal self-mask (uniform branch: fires 1/128 iterations per wave).
    if (kb + 32 > q0 && kb <= q0 + 15) {
#pragma unroll
      for (int r = 0; r < 8; ++r) {
        const int m = q0 + r + hi * 8;
        if (m == kb + lo)      s0[r] = -3.0e38f;
        if (m == kb + 16 + lo) s1[r] = -3.0e38f;
      }
    }

    // Row max across the 16-key tile pair (butterfly over 16-lane row groups)
    float tm[8];
#pragma unroll
    for (int r = 0; r < 8; ++r) tm[r] = fmaxf(s0[r], s1[r]);
#pragma unroll
    for (int xm = 1; xm <= 8; xm <<= 1)
#pragma unroll
      for (int r = 0; r < 8; ++r) tm[r] = fmaxf(tm[r], __shfl_xor(tm[r], xm, 32));

    float mnew[8], corr[8], ts[8];
    f32x8 p0, p1;
#pragma unroll
    for (int r = 0; r < 8; ++r) {
      mnew[r] = fmaxf(mi[r], tm[r]);
      p0[r] = exp2f((s0[r] - mnew[r]) * L2E);
      p1[r] = exp2f((s1[r] - mnew[r]) * L2E);
      ts[r] = p0[r] + p1[r];
    }
#pragma unroll
    for (int xm = 1; xm <= 8; xm <<= 1)
#pragma unroll
      for (int r = 0; r < 8; ++r) ts[r] += __shfl_xor(ts[r], xm, 32);
#pragma unroll
    for (int r = 0; r < 8; ++r) {
      corr[r] = exp2f((mi[r] - mnew[r]) * L2E);
      li[r] = li[r] * corr[r] + ts[r];
      mi[r] = mnew[r];
    }

    // P: C-layout -> LDS -> A-layout (per-wave tile; DScnt orders RAW).
#pragma unroll
    for (int r = 0; r < 8; ++r) {
      const int m = r + hi * 8;
      P[wave][m][lo]      = (__bf16)p0[r];
      P[wave][m][16 + lo] = (__bf16)p1[r];
    }
    AFrag pa;
    {
      const __bf16* pr = &P[wave][lo][0];
      pa.h[0] = *(const bf16x8*)(pr + hi * 16);
      pa.h[1] = *(const bf16x8*)(pr + 8 + hi * 16);
    }

    // O = O*corr + P @ V  (V B-fragments contiguous thanks to Vt layout)
#pragma unroll
    for (int nt = 0; nt < 4; ++nt) {
#pragma unroll
      for (int r = 0; r < 8; ++r) O[nt][r] *= corr[r];
      bf16x16 vf = *(const bf16x16*)(Vth + (size_t)(nt * 16 + lo) * N_ + kb + hi * 16);
      O[nt] = wmma_bf16(pa.v, vf, O[nt]);
    }
  }

  float inv[8];
#pragma unroll
  for (int r = 0; r < 8; ++r) inv[r] = 1.0f / li[r];
#pragma unroll
  for (int nt = 0; nt < 4; ++nt)
#pragma unroll
    for (int r = 0; r < 8; ++r) {
      const int m = q0 + r + hi * 8;
      Ob[headOff + (size_t)m * HD_ + nt * 16 + lo] = (__bf16)(O[nt][r] * inv[r]);
    }
}

// ---------------------------------------------------------------------------
// Kernel 3: Y = Ob (4096x512) @ Wout (512x512) + b_out, fp32 output.
// Block tile 128x128, wave tile 32x64 like qkv_gemm.
// ---------------------------------------------------------------------------
__global__ __launch_bounds__(256) void out_gemm(
    const __bf16* __restrict__ Ob, const __bf16* __restrict__ Wt,
    const float* __restrict__ bias, float* __restrict__ Y)
{
  const int lane = threadIdx.x & 31;
  const int wave = threadIdx.x >> 5;
  const int lo = lane & 15, hi = lane >> 4;
  const int b = blockIdx.z;
  const int row0    = blockIdx.x * 128 + (wave & 3) * 32;
  const int colbase = blockIdx.y * 128 + (wave >> 2) * 64;
  const __bf16* or0 = Ob + ((size_t)b * N_ + row0 + lo) * DIM_;
  const __bf16* or1 = Ob + ((size_t)b * N_ + row0 + 16 + lo) * DIM_;

  f32x8 acc[2][4];
#pragma unroll
  for (int i = 0; i < 2; ++i)
#pragma unroll
    for (int nt = 0; nt < 4; ++nt)
#pragma unroll
      for (int r = 0; r < 8; ++r) acc[i][nt][r] = 0.0f;

#pragma unroll 2
  for (int kc = 0; kc < DIM_; kc += 32) {
    AFrag a0, a1;
    a0.h[0] = *(const bf16x8*)(or0 + kc + hi * 16);
    a0.h[1] = *(const bf16x8*)(or0 + kc + 8 + hi * 16);
    a1.h[0] = *(const bf16x8*)(or1 + kc + hi * 16);
    a1.h[1] = *(const bf16x8*)(or1 + kc + 8 + hi * 16);
#pragma unroll
    for (int nt = 0; nt < 4; ++nt) {
      bf16x16 bf = *(const bf16x16*)(Wt + (size_t)(colbase + nt * 16 + lo) * DIM_
                                        + kc + hi * 16);
      acc[0][nt] = wmma_bf16(a0.v, bf, acc[0][nt]);
      acc[1][nt] = wmma_bf16(a1.v, bf, acc[1][nt]);
    }
  }

#pragma unroll
  for (int i = 0; i < 2; ++i)
#pragma unroll
  for (int nt = 0; nt < 4; ++nt) {
    const int c = colbase + nt * 16 + lo;
    const float bv = bias[c];
#pragma unroll
    for (int r = 0; r < 8; ++r) {
      const int n = row0 + i * 16 + r + hi * 8;
      Y[((size_t)b * N_ + n) * DIM_ + c] = acc[i][nt][r] + bv;
    }
  }
}

// ---------------------------------------------------------------------------
extern "C" void kernel_launch(void* const* d_in, const int* in_sizes, int n_in,
                              void* d_out, int out_size, void* d_ws, size_t ws_size,
                              hipStream_t stream) {
  (void)in_sizes; (void)n_in; (void)out_size; (void)ws_size;

  const float* x    = (const float*)d_in[0];
  const float* wqkv = (const float*)d_in[1];
  const float* temp = (const float*)d_in[2];
  const float* wout = (const float*)d_in[3];
  const float* bout = (const float*)d_in[4];
  float* y = (float*)d_out;

  // Workspace carve-up (bf16 elements): ~44 MB total, fits comfortably in L2.
  __bf16* p      = (__bf16*)d_ws;
  __bf16* xbf    = p; p += (size_t)B_ * N_ * DIM_;     // 4,194,304
  __bf16* wqkvT  = p; p += (size_t)QKVC_ * DIM_;       //   786,432
  __bf16* woutT  = p; p += (size_t)DIM_ * DIM_;        //   262,144
  __bf16* Qb     = p; p += (size_t)B_ * N_ * 512;      // 4,194,304
  __bf16* Kb     = p; p += (size_t)B_ * N_ * 512;
  __bf16* Vt     = p; p += (size_t)B_ * N_ * 512;
  __bf16* Ob     = p; p += (size_t)B_ * N_ * 512;

  prep_kernel<<<2048, 256, 0, stream>>>(x, wqkv, wout, xbf, wqkvT, woutT);
  qkv_gemm<<<dim3(32, 12, 2), 256, 0, stream>>>(xbf, wqkvT, Qb, Kb, Vt);
  attn_kernel<<<dim3(32, 8, 2), 256, 0, stream>>>(Qb, Kb, Vt, temp, Ob);
  out_gemm<<<dim3(32, 4, 2), 256, 0, stream>>>(Ob, woutT, bout, y);
}
